// orderImageNet_9543417332117
// MI455X (gfx1250) — matrix operand order repaired and checked by reading, not compiled, not run on hardware
//
#include <hip/hip_runtime.h>

typedef __bf16 bf16_t;
typedef __attribute__((ext_vector_type(16))) __bf16 v16bf;
typedef __attribute__((ext_vector_type(8)))  __bf16 v8bf;
typedef __attribute__((ext_vector_type(8)))  float   v8f;

#define BATCH 32768
#define NIMG  5
#define INDIM 512
#define DPAD  608   // 600 padded to multiple of 16/32
#define CPAD  224   // 200 conv channels padded to 14 tiles (7 tile-pairs)

// ---------------------------------------------------------------------------
// WMMA fragment helpers (wave32, 16x16x32 bf16)
// ---------------------------------------------------------------------------
static __device__ __forceinline__ v8f wmma_bf16(v16bf a, v16bf b, v8f c) {
  return __builtin_amdgcn_wmma_f32_16x16x32_bf16(false, a, false, b, (short)0, c,
                                                 false, false);
}

static __device__ __forceinline__ v16bf load_a_from_bf16(const bf16_t* rowPtr, int k0) {
  const int half = (threadIdx.x >> 4) & 1;
  const bf16_t* p = rowPtr + k0 + half * 8;
  v8bf lo = *(const v8bf*)(p);
  v8bf hi = *(const v8bf*)(p + 16);
  v16bf f;
#pragma unroll
  for (int e = 0; e < 8; ++e) { f[e] = lo[e]; f[e + 8] = hi[e]; }
  return f;
}

static __device__ __forceinline__ v16bf load_a_from_f32(const float* rowPtr, int k0) {
  const int half = (threadIdx.x >> 4) & 1;
  const float* p = rowPtr + k0 + half * 8;
  v8f lo = *(const v8f*)(p);
  v8f hi = *(const v8f*)(p + 16);
  v16bf f;
#pragma unroll
  for (int e = 0; e < 8; ++e) { f[e] = (__bf16)lo[e]; f[e + 8] = (__bf16)hi[e]; }
  return f;
}

static __device__ __forceinline__ v16bf load_b_frag(const bf16_t* btBase,
                                                    int strideElems, int k0) {
  const int lane = threadIdx.x & 31;
  const int c = lane & 15, half = lane >> 4;
  const bf16_t* p = btBase + (size_t)c * strideElems + k0 + half * 8;
  v8bf lo = *(const v8bf*)(p);
  v8bf hi = *(const v8bf*)(p + 16);
  v16bf f;
#pragma unroll
  for (int e = 0; e < 8; ++e) { f[e] = lo[e]; f[e + 8] = hi[e]; }
  return f;
}

// ---------------------------------------------------------------------------
// Prep kernels
// ---------------------------------------------------------------------------
__global__ void k_cvt_pad_rows(const float* __restrict__ src, bf16_t* __restrict__ dst,
                               int R0, int Rp, int C) {
  int idx = blockIdx.x * blockDim.x + threadIdx.x;
  if (idx >= Rp * C) return;
  int rr = idx / C, cc = idx % C;
  dst[idx] = (rr < R0) ? (bf16_t)src[(size_t)rr * C + cc] : (bf16_t)0.0f;
}

// src is [K0][N0] (in-major); dst is [Np][Kp] (out-major, zero padded)
__global__ void k_transpose_cvt(const float* __restrict__ src, bf16_t* __restrict__ dst,
                                int N0, int K0, int Np, int Kp) {
  int idx = blockIdx.x * blockDim.x + threadIdx.x;
  if (idx >= Np * Kp) return;
  int nn = idx / Kp, kk = idx % Kp;
  dst[idx] = (nn < N0 && kk < K0) ? (bf16_t)src[(size_t)kk * N0 + nn] : (bf16_t)0.0f;
}

__global__ void k_pad_f32(const float* __restrict__ src, float* __restrict__ dst,
                          int n0, int np) {
  int i = blockIdx.x * blockDim.x + threadIdx.x;
  if (i >= np) return;
  dst[i] = (i < n0) ? src[i] : 0.0f;
}

// ---------------------------------------------------------------------------
// Conv blocks as WMMA GEMMs + bias + relu + max over t.
// grid (2048, 3), 256 thr. wave unit u = conv*7 + tilePair (21 units).
// One A fragment feeds TWO channel tiles (halves A loads + f32->bf16 cvt).
// ---------------------------------------------------------------------------
__global__ __launch_bounds__(256) void k_conv(const float* __restrict__ feats,
    const bf16_t* __restrict__ w2b, const bf16_t* __restrict__ w3b,
    const bf16_t* __restrict__ w5b, const float* __restrict__ biasc,
    bf16_t* __restrict__ cbuf) {
  const int wave = threadIdx.x >> 5;
  const int unit = blockIdx.y * 8 + wave;
  if (unit >= 21) return;
  const int conv = unit / 7;
  const int ct0 = (unit % 7) * 2;            // first of two channel tiles
  const int Ksz = (conv == 0) ? 2 : (conv == 1) ? 3 : 5;
  const int T   = 6 - Ksz;                   // 4, 3, 1
  const int KK  = Ksz * INDIM;               // 1024, 1536, 2560
  const bf16_t* W = (conv == 0) ? w2b : (conv == 1) ? w3b : w5b;
  const bf16_t* bb0 = W + (size_t)(ct0 * 16) * KK;
  const bf16_t* bb1 = bb0 + (size_t)16 * KK;

  const int lane = threadIdx.x & 31;
  const int r = lane & 15, half = lane >> 4;
  const int s0 = blockIdx.x * 16;
  const float* featRow = feats + (size_t)(s0 + r) * (NIMG * INDIM);
  const float bias0 = biasc[conv * CPAD + ct0 * 16 + r];
  const float bias1 = biasc[conv * CPAD + ct0 * 16 + 16 + r];

  float m0v[8], m1v[8];
#pragma unroll
  for (int j = 0; j < 8; ++j) { m0v[j] = 0.0f; m1v[j] = 0.0f; }

  for (int t = 0; t < T; ++t) {
    v8f acc0 = {}, acc1 = {};
    const float* arow = featRow + t * INDIM;   // window [t, t+Ksz) is contiguous
    for (int k0 = 0; k0 < KK; k0 += 32) {
      v16bf a  = load_a_from_f32(arow, k0);
      v16bf b0 = load_b_frag(bb0, KK, k0);
      v16bf b1 = load_b_frag(bb1, KK, k0);
      acc0 = wmma_bf16(a, b0, acc0);
      acc1 = wmma_bf16(a, b1, acc1);
    }
#pragma unroll
    for (int j = 0; j < 8; ++j) {
      m0v[j] = fmaxf(m0v[j], acc0[j] + bias0);
      m1v[j] = fmaxf(m1v[j], acc1[j] + bias1);
    }
  }

  const int chMax = (conv == 2) ? 208 : 200;   // conv2 pad chans zero cols 600..607
#pragma unroll
  for (int tile = 0; tile < 2; ++tile) {
    const int ch = ct0 * 16 + tile * 16 + r;
    if (ch < chMax) {
      const int col = conv * 200 + ch;
      const float* mv = tile ? m1v : m0v;
#pragma unroll
      for (int j = 0; j < 8; ++j)
        cbuf[(size_t)(s0 + half * 8 + j) * DPAD + col] = (bf16_t)fmaxf(mv[j], 0.0f);
    }
  }
}

// ---------------------------------------------------------------------------
// Generic bf16 WMMA GEMM with 2M x 2N register tiling: 4 WMMAs per
// (2 A-frag + 2 B-frag) loads -> 1 load per WMMA.
// grid (M/32, ceil(Npairs/8)), 256 thr, one 32x32 macro-tile per wave.
// ---------------------------------------------------------------------------
template <bool RELU>
__global__ __launch_bounds__(256) void k_gemm(const bf16_t* __restrict__ A, int lda,
    const bf16_t* __restrict__ BT, int Ka, const float* __restrict__ bias,
    bf16_t* __restrict__ C, int ldc, int Npairs) {
  const int wave = threadIdx.x >> 5;
  const int np = blockIdx.y * 8 + wave;
  if (np >= Npairs) return;
  const int lane = threadIdx.x & 31;
  const int r = lane & 15, half = lane >> 4;
  const int m0 = blockIdx.x * 32;
  const bf16_t* arow0 = A + (size_t)(m0 + r) * lda;
  const bf16_t* arow1 = A + (size_t)(m0 + 16 + r) * lda;
  const bf16_t* bb0 = BT + (size_t)(np * 32) * Ka;
  const bf16_t* bb1 = bb0 + (size_t)16 * Ka;

  v8f acc00 = {}, acc01 = {}, acc10 = {}, acc11 = {};
  for (int k0 = 0; k0 < Ka; k0 += 32) {
    v16bf a0 = load_a_from_bf16(arow0, k0);
    v16bf a1 = load_a_from_bf16(arow1, k0);
    v16bf b0 = load_b_frag(bb0, Ka, k0);
    v16bf b1 = load_b_frag(bb1, Ka, k0);
    acc00 = wmma_bf16(a0, b0, acc00);
    acc01 = wmma_bf16(a0, b1, acc01);
    acc10 = wmma_bf16(a1, b0, acc10);
    acc11 = wmma_bf16(a1, b1, acc11);
  }
  const float bv0 = bias[np * 32 + r];
  const float bv1 = bias[np * 32 + 16 + r];
#pragma unroll
  for (int j = 0; j < 8; ++j) {
    float v00 = acc00[j] + bv0, v01 = acc01[j] + bv1;
    float v10 = acc10[j] + bv0, v11 = acc11[j] + bv1;
    if (RELU) {
      v00 = fmaxf(v00, 0.0f); v01 = fmaxf(v01, 0.0f);
      v10 = fmaxf(v10, 0.0f); v11 = fmaxf(v11, 0.0f);
    }
    const size_t row0 = (size_t)(m0 + half * 8 + j) * ldc;
    const size_t row1 = (size_t)(m0 + 16 + half * 8 + j) * ldc;
    C[row0 + np * 32 + r]      = (bf16_t)v00;
    C[row0 + np * 32 + 16 + r] = (bf16_t)v01;
    C[row1 + np * 32 + r]      = (bf16_t)v10;
    C[row1 + np * 32 + 16 + r] = (bf16_t)v11;
  }
}

// ---------------------------------------------------------------------------
// fc0 (split-K over [feats | ci]) + relu + fcs dot -> score, fully fused.
// 2M x 2N register tiling; rows = (n, i) flat; M-tile 32 per block.
// grid (163840/32), 256 thr; each wave does 2 N-tile-pairs.
// ---------------------------------------------------------------------------
__global__ __launch_bounds__(256) void k_fc0_score(const float* __restrict__ feats,
    const bf16_t* __restrict__ ci, const bf16_t* __restrict__ fc0t,
    const float* __restrict__ fc0_b, const float* __restrict__ fcs_w,
    const float* __restrict__ fcs_b, float* __restrict__ score) {
  __shared__ float sbuf[32];
  const int tid = threadIdx.x;
  if (tid < 32) sbuf[tid] = 0.0f;
  __syncthreads();
  const int wave = tid >> 5, lane = tid & 31;
  const int r = lane & 15, half = lane >> 4;
  const int m0 = blockIdx.x * 32;
  const int row0 = m0 + r, row1 = m0 + 16 + r;   // row = n*5 + i
  const float*  frow0 = feats + (size_t)row0 * INDIM;
  const float*  frow1 = feats + (size_t)row1 * INDIM;
  const bf16_t* crow0 = ci + (size_t)(row0 / NIMG) * INDIM;
  const bf16_t* crow1 = ci + (size_t)(row1 / NIMG) * INDIM;

  float part0[8], part1[8];
#pragma unroll
  for (int j = 0; j < 8; ++j) { part0[j] = 0.0f; part1[j] = 0.0f; }

  for (int np = wave; np < 16; np += 8) {        // 16 N-tile-pairs (512 hidden)
    v8f acc00 = {}, acc01 = {}, acc10 = {}, acc11 = {};
    const bf16_t* bb0 = fc0t + (size_t)(np * 32) * 1024;
    const bf16_t* bb1 = bb0 + (size_t)16 * 1024;
    for (int k0 = 0; k0 < INDIM; k0 += 32) {     // K phase 1: per-image feature
      v16bf a0 = load_a_from_f32(frow0, k0);
      v16bf a1 = load_a_from_f32(frow1, k0);
      v16bf b0 = load_b_frag(bb0, 1024, k0);
      v16bf b1 = load_b_frag(bb1, 1024, k0);
      acc00 = wmma_bf16(a0, b0, acc00);
      acc01 = wmma_bf16(a0, b1, acc01);
      acc10 = wmma_bf16(a1, b0, acc10);
      acc11 = wmma_bf16(a1, b1, acc11);
    }
    for (int k0 = 0; k0 < INDIM; k0 += 32) {     // K phase 2: broadcast context
      v16bf a0 = load_a_from_bf16(crow0, k0);
      v16bf a1 = load_a_from_bf16(crow1, k0);
      v16bf b0 = load_b_frag(bb0, 1024, INDIM + k0);
      v16bf b1 = load_b_frag(bb1, 1024, INDIM + k0);
      acc00 = wmma_bf16(a0, b0, acc00);
      acc01 = wmma_bf16(a0, b1, acc01);
      acc10 = wmma_bf16(a1, b0, acc10);
      acc11 = wmma_bf16(a1, b1, acc11);
    }
    const int col0 = np * 32 + r, col1 = col0 + 16;
    const float bbc0 = fc0_b[col0], wsc0 = fcs_w[col0];
    const float bbc1 = fc0_b[col1], wsc1 = fcs_w[col1];
#pragma unroll
    for (int j = 0; j < 8; ++j) {
      part0[j] += fmaxf(acc00[j] + bbc0, 0.0f) * wsc0
                + fmaxf(acc01[j] + bbc1, 0.0f) * wsc1;
      part1[j] += fmaxf(acc10[j] + bbc0, 0.0f) * wsc0
                + fmaxf(acc11[j] + bbc1, 0.0f) * wsc1;
    }
  }

  // reduce across 16 lanes of each half (cols), then across waves via LDS
#pragma unroll
  for (int j = 0; j < 8; ++j) {
#pragma unroll
    for (int s = 1; s < 16; s <<= 1) {
      part0[j] += __shfl_xor(part0[j], s, 32);
      part1[j] += __shfl_xor(part1[j], s, 32);
    }
    if (r == 0) {
      atomicAdd(&sbuf[half * 8 + j], part0[j]);
      atomicAdd(&sbuf[16 + half * 8 + j], part1[j]);
    }
  }
  __syncthreads();
  if (tid < 32) score[m0 + tid] = sbuf[tid] + fcs_b[0];
}

// ---------------------------------------------------------------------------
// rank under stable descending sort (double argsort) -> int64
// ---------------------------------------------------------------------------
__global__ void k_order(const float* __restrict__ score, long long* __restrict__ order,
                        int n) {
  int i = blockIdx.x * blockDim.x + threadIdx.x;
  if (i >= n) return;
  float s[5];
#pragma unroll
  for (int k = 0; k < 5; ++k) s[k] = score[i * 5 + k];
#pragma unroll
  for (int k = 0; k < 5; ++k) {
    int rank = 0;
#pragma unroll
    for (int j = 0; j < 5; ++j)
      rank += ((s[j] > s[k]) || ((s[j] == s[k]) && (j < k))) ? 1 : 0;
    order[i * 5 + k] = (long long)rank;
  }
}

// ---------------------------------------------------------------------------
extern "C" void kernel_launch(void* const* d_in, const int* in_sizes, int n_in,
                              void* d_out, int out_size, void* d_ws, size_t ws_size,
                              hipStream_t stream) {
  (void)in_sizes; (void)n_in; (void)out_size; (void)ws_size;
  const float* feats = (const float*)d_in[0];
  const float* w2    = (const float*)d_in[1];
  const float* b2    = (const float*)d_in[2];
  const float* w3    = (const float*)d_in[3];
  const float* b3    = (const float*)d_in[4];
  const float* w5    = (const float*)d_in[5];
  const float* b5    = (const float*)d_in[6];
  const float* fc1_w = (const float*)d_in[7];
  const float* fc1_b = (const float*)d_in[8];
  const float* fc2_w = (const float*)d_in[9];
  const float* fc2_b = (const float*)d_in[10];
  const float* fc0_w = (const float*)d_in[11];
  const float* fc0_b = (const float*)d_in[12];
  const float* fcs_w = (const float*)d_in[13];
  const float* fcs_b = (const float*)d_in[14];

  char* ws = (char*)d_ws;
  size_t off = 0;
  auto alloc = [&](size_t bytes) -> char* {
    char* p = ws + off;
    off += (bytes + 255) & ~(size_t)255;
    return p;
  };
  bf16_t* w2b   = (bf16_t*)alloc((size_t)CPAD * 1024 * 2);
  bf16_t* w3b   = (bf16_t*)alloc((size_t)CPAD * 1536 * 2);
  bf16_t* w5b   = (bf16_t*)alloc((size_t)CPAD * 2560 * 2);
  bf16_t* fc1t  = (bf16_t*)alloc((size_t)DPAD * DPAD * 2);
  bf16_t* fc2t  = (bf16_t*)alloc((size_t)INDIM * DPAD * 2);
  bf16_t* fc0t  = (bf16_t*)alloc((size_t)INDIM * 1024 * 2);
  float*  biasc = (float*)alloc(3 * CPAD * 4);
  float*  fc1bp = (float*)alloc(DPAD * 4);
  bf16_t* cbuf  = (bf16_t*)alloc((size_t)BATCH * DPAD * 2);
  bf16_t* hbuf  = (bf16_t*)alloc((size_t)BATCH * DPAD * 2);
  bf16_t* cibuf = (bf16_t*)alloc((size_t)BATCH * INDIM * 2);

  auto cdiv = [](int a, int b) { return (a + b - 1) / b; };

  // ---- weight prep (bf16 convert / pad / transpose to out-major) ----
  k_cvt_pad_rows<<<cdiv(CPAD * 1024, 256), 256, 0, stream>>>(w2, w2b, 200, CPAD, 1024);
  k_cvt_pad_rows<<<cdiv(CPAD * 1536, 256), 256, 0, stream>>>(w3, w3b, 200, CPAD, 1536);
  k_cvt_pad_rows<<<cdiv(CPAD * 2560, 256), 256, 0, stream>>>(w5, w5b, 200, CPAD, 2560);
  k_transpose_cvt<<<cdiv(DPAD * DPAD, 256), 256, 0, stream>>>(fc1_w, fc1t, 600, 600, DPAD, DPAD);
  k_transpose_cvt<<<cdiv(INDIM * DPAD, 256), 256, 0, stream>>>(fc2_w, fc2t, INDIM, 600, INDIM, DPAD);
  k_transpose_cvt<<<cdiv(INDIM * 1024, 256), 256, 0, stream>>>(fc0_w, fc0t, INDIM, 1024, INDIM, 1024);
  k_pad_f32<<<1, 256, 0, stream>>>(b2, biasc + 0 * CPAD, 200, CPAD);
  k_pad_f32<<<1, 256, 0, stream>>>(b3, biasc + 1 * CPAD, 200, CPAD);
  k_pad_f32<<<1, 256, 0, stream>>>(b5, biasc + 2 * CPAD, 200, CPAD);
  k_pad_f32<<<3, 256, 0, stream>>>(fc1_b, fc1bp, 600, DPAD);

  // ---- pipeline ----
  k_conv<<<dim3(BATCH / 16, 3), 256, 0, stream>>>(feats, w2b, w3b, w5b, biasc, cbuf);
  k_gemm<true ><<<dim3(BATCH / 32, 3), 256, 0, stream>>>(cbuf, DPAD, fc1t, DPAD, fc1bp,
                                                         hbuf, DPAD, 19);
  k_gemm<false><<<dim3(BATCH / 32, 2), 256, 0, stream>>>(hbuf, DPAD, fc2t, DPAD, fc2_b,
                                                         cibuf, INDIM, 16);

  float* score = (float*)d_out;
  long long* order = (long long*)((char*)d_out + (size_t)BATCH * NIMG * sizeof(float));
  k_fc0_score<<<dim3(BATCH * NIMG / 32), 256, 0, stream>>>(feats, cibuf, fc0t, fc0_b,
                                                           fcs_w, fcs_b, score);
  k_order<<<cdiv(BATCH, 256), 256, 0, stream>>>(score, order, BATCH);
}
